// EncoderBlock_687194767983
// MI455X (gfx1250) — matrix operand order repaired and checked by reading, not compiled, not run on hardware
//
#include <hip/hip_runtime.h>
#include <math.h>

typedef __bf16 bf16_t;
typedef __attribute__((ext_vector_type(16))) __bf16 v16bf;
typedef __attribute__((ext_vector_type(8)))  __bf16 v8bf;
typedef __attribute__((ext_vector_type(8)))  float  v8f;
typedef __attribute__((ext_vector_type(4)))  unsigned int v4u;
typedef __attribute__((ext_vector_type(8)))  int v8i;
typedef __attribute__((ext_vector_type(4)))  int v4i;

union FragAB { v16bf v; v8bf h[2]; };
union FragC  { v8f   v; float f[8]; };

__device__ inline v8f wmma_bf16(v16bf a, v16bf b, v8f c) {
  // D = A(16x32) * B(32x16) + C, fp32 accumulate
  return __builtin_amdgcn_wmma_f32_16x16x32_bf16(false, a, false, b, (short)0, c, false, false);
}

__device__ inline bf16_t f2bf(float x) { return (bf16_t)x; }

// ---------------------------------------------------------------- conversions
__global__ void cvt_f32_to_bf16(const float* __restrict__ in, bf16_t* __restrict__ out, int n) {
  int i = blockIdx.x * blockDim.x + threadIdx.x;
  if (i < n) out[i] = f2bf(in[i]);
}

// in: [K][N] row-major fp32  ->  out: [N][K] bf16 (transposed)
__global__ void cvt_transpose_f32_to_bf16(const float* __restrict__ in, bf16_t* __restrict__ out,
                                          int K, int N) {
  int i = blockIdx.x * blockDim.x + threadIdx.x;
  if (i < K * N) {
    int k = i / N, n = i - k * N;
    out[(size_t)n * K + k] = f2bf(in[i]);
  }
}

// ---------------------------------------------------------------- TDM staging
#define HAS_TDM __has_builtin(__builtin_amdgcn_tensor_load_to_lds)

#if HAS_TDM
// Issue a TDM 2D tile load: tile = 128 rows x 64 bf16, row stride in LDS padded
// to 72 elements via the D# pad feature (pad_interval=32 DW -> code 4,
// pad_amount=4 DW -> code 3). One instruction per wave; tracked by TENSORcnt.
__device__ inline void tdm_load_tile(const bf16_t* g, unsigned lds_off,
                                     int tensor_d0, int tensor_d1, long stride0) {
  unsigned long long ga = (unsigned long long)(uintptr_t)g;
  v4u g0;
  g0[0] = 1u;                                        // count=1 (valid), user mode
  g0[1] = lds_off;                                   // LDS byte address
  g0[2] = (unsigned)ga;                              // global_addr[31:0]
  g0[3] = (unsigned)((ga >> 32) & 0x01ffffffull) | (2u << 30); // addr[56:32] | type=2
  unsigned td0 = (unsigned)tensor_d0, td1 = (unsigned)tensor_d1;
  v8i g1;
  g1[0] = (int)((1u << 16)      // data_size = 2 bytes
              | (1u << 20)      // pad_enable
              | (4u << 22)      // pad_interval = 32 DWORDs (one 128B row)
              | (3u << 25));    // pad_amount   = 4 DWORDs (16B)
  g1[1] = (int)((td0 & 0xffffu) << 16);                     // tensor_dim0[15:0]
  g1[2] = (int)((td0 >> 16) | ((td1 & 0xffffu) << 16));     // td0[31:16] | td1[15:0]
  g1[3] = (int)((td1 >> 16) | (64u << 16));                 // td1[31:16] | tile_dim0=64
  g1[4] = 128;                                              // tile_dim1=128, tile_dim2=0
  g1[5] = (int)((unsigned long long)stride0 & 0xffffffffull);   // dim0_stride[31:0]
  g1[6] = (int)(((unsigned long long)stride0 >> 32) & 0xffffull); // stride[47:32]
  g1[7] = 0;                                                // dim1_stride unused (2D)
  v4i z = (v4i)0;
#if __clang_major__ >= 23
  v8i z8 = (v8i)0;
  __builtin_amdgcn_tensor_load_to_lds(g0, g1, z, z, z8, 0);
#else
  __builtin_amdgcn_tensor_load_to_lds(g0, g1, z, z, 0);
#endif
}
#endif

// ---------------------------------------------------------------- WMMA GEMM
// C[M][N] = act(A[M][K] * Bt[N][K]^T + bias), A/Bt bf16, C bf16, fp32 accum.
#define BM 128
#define BN 128
#define BKK 64
#define LDT 72   // padded LDS stride (elements); 144B rows keep 16B alignment

__global__ void __launch_bounds__(256)
gemm_bf16_wmma(const bf16_t* __restrict__ A, const bf16_t* __restrict__ Bt,
               const float* __restrict__ bias, bf16_t* __restrict__ C,
               int M, int N, int K, int gelu) {
  __shared__ bf16_t As[2][BM][LDT];
  __shared__ bf16_t Bs[2][BM][LDT];

  const int tid  = threadIdx.x;
  const int lane = tid & 31;
  const int wave = tid >> 5;
  const int half = lane >> 4;
  const int l16  = lane & 15;

  const int blockN0 = blockIdx.x * BN;
  const int blockM0 = blockIdx.y * BM;
  const int waveM   = (wave >> 2) * 64;   // 0 / 64
  const int waveN   = (wave & 3) * 32;    // 0 / 32 / 64 / 96

  FragC acc[4][2];
  #pragma unroll
  for (int i = 0; i < 4; i++)
    #pragma unroll
    for (int j = 0; j < 2; j++)
      #pragma unroll
      for (int r = 0; r < 8; r++) acc[i][j].f[r] = 0.0f;

#if HAS_TDM
  const bool issuer = (wave == 0);
  if (issuer) {
    tdm_load_tile(A  + (size_t)blockM0 * K, (unsigned)(uintptr_t)&As[0][0][0], K, M, K);
    tdm_load_tile(Bt + (size_t)blockN0 * K, (unsigned)(uintptr_t)&Bs[0][0][0], K, N, K);
    __builtin_amdgcn_s_wait_tensorcnt(0);
  }
  __syncthreads();
#endif

  int buf = 0;
  for (int k0 = 0; k0 < K; k0 += BKK) {
#if HAS_TDM
    // prefetch next K-tile into the other buffer while we compute on this one
    if (k0 + BKK < K && issuer) {
      tdm_load_tile(A  + (size_t)blockM0 * K + (k0 + BKK),
                    (unsigned)(uintptr_t)&As[buf ^ 1][0][0], K, M, K);
      tdm_load_tile(Bt + (size_t)blockN0 * K + (k0 + BKK),
                    (unsigned)(uintptr_t)&Bs[buf ^ 1][0][0], K, N, K);
    }
#else
    __syncthreads();
    #pragma unroll
    for (int l = 0; l < 4; l++) {                  // 1024 16B-loads per operand tile
      int flat = tid + l * 256;                    // 0..1023
      int r = flat >> 3;                           // 0..127
      int c = (flat & 7) << 3;                     // 0,8,..,56
      *(v8bf*)&As[0][r][c] = *(const v8bf*)&A [(size_t)(blockM0 + r) * K + k0 + c];
      *(v8bf*)&Bs[0][r][c] = *(const v8bf*)&Bt[(size_t)(blockN0 + r) * K + k0 + c];
    }
    __syncthreads();
#endif

    #pragma unroll
    for (int kk = 0; kk < BKK; kk += 32) {
      FragAB a[4], b[2];
      #pragma unroll
      for (int i = 0; i < 4; i++) {                // A frag: row = l16, K split 8*half / 16+8*half
        const bf16_t* p = &As[buf][waveM + i * 16 + l16][kk];
        a[i].h[0] = *(const v8bf*)(p + 8 * half);
        a[i].h[1] = *(const v8bf*)(p + 16 + 8 * half);
      }
      #pragma unroll
      for (int j = 0; j < 2; j++) {                // B frag: col = l16, K split 8*half / 16+8*half
        const bf16_t* p = &Bs[buf][waveN + j * 16 + l16][kk];
        b[j].h[0] = *(const v8bf*)(p + 8 * half);
        b[j].h[1] = *(const v8bf*)(p + 16 + 8 * half);
      }
      #pragma unroll
      for (int i = 0; i < 4; i++)
        #pragma unroll
        for (int j = 0; j < 2; j++)
          acc[i][j].v = wmma_bf16(a[i].v, b[j].v, acc[i][j].v);
    }

#if HAS_TDM
    if (k0 + BKK < K) {
      if (issuer) __builtin_amdgcn_s_wait_tensorcnt(0);
      __syncthreads();
      buf ^= 1;
    }
#endif
  }

  #pragma unroll
  for (int i = 0; i < 4; i++)
    #pragma unroll
    for (int j = 0; j < 2; j++) {
      int col = blockN0 + waveN + j * 16 + l16;
      float bv = bias[col];
      #pragma unroll
      for (int r = 0; r < 8; r++) {
        int row = blockM0 + waveM + i * 16 + half * 8 + r;
        float x = acc[i][j].f[r] + bv;
        if (gelu) x = 0.5f * x * (1.0f + erff(x * 0.70710678f));
        C[(size_t)row * N + col] = f2bf(x);
      }
    }
}

// ---------------------------------------------------------------- flash attention
// Q,K,V,O: [B*T][H*64] bf16. One block = 64 query rows of one (batch, head).
#define FA_LD 72

__global__ void __launch_bounds__(128)
flash_attn_wmma(const bf16_t* __restrict__ Q, const bf16_t* __restrict__ Km,
                const bf16_t* __restrict__ Vm, const int* __restrict__ mask,
                bf16_t* __restrict__ O, int T, int nheads) {
  __shared__ bf16_t Ks[64][FA_LD];        // [key][d]
  __shared__ bf16_t Vs[64][FA_LD];        // [d][key] (transposed)
  __shared__ bf16_t Ps[4][16][FA_LD];     // per-wave P staging [row][key]

  const int tid  = threadIdx.x;
  const int lane = tid & 31;
  const int wave = tid >> 5;
  const int half = lane >> 4;
  const int l16  = lane & 15;

  const int qblk  = blockIdx.x;
  const int head  = blockIdx.y;
  const int batch = blockIdx.z;
  const size_t rs = (size_t)nheads * 64;  // 1024

  const bf16_t* Qb = Q  + (size_t)batch * T * rs + head * 64;
  const bf16_t* Kb = Km + (size_t)batch * T * rs + head * 64;
  const bf16_t* Vb = Vm + (size_t)batch * T * rs + head * 64;
  bf16_t*       Ob = O  + (size_t)batch * T * rs + head * 64;
  const int*    mb = mask + batch * T;

  // Q fragments (16 rows per wave), registers for whole kernel
  const int qrow = qblk * 64 + wave * 16 + l16;
  FragAB qa[2];
  #pragma unroll
  for (int c = 0; c < 2; c++) {
    const bf16_t* p = Qb + (size_t)qrow * rs + c * 32;
    qa[c].h[0] = *(const v8bf*)(p + 8 * half);
    qa[c].h[1] = *(const v8bf*)(p + 16 + 8 * half);
  }

  FragC o[4];
  float rm[8], rl[8];
  #pragma unroll
  for (int t = 0; t < 4; t++)
    #pragma unroll
    for (int r = 0; r < 8; r++) o[t].f[r] = 0.0f;
  #pragma unroll
  for (int r = 0; r < 8; r++) { rm[r] = -INFINITY; rl[r] = 0.0f; }

  for (int kb = 0; kb < T; kb += 64) {
    __syncthreads();
    #pragma unroll
    for (int l = 0; l < 4; l++) {
      int flat = tid + l * 128;            // 0..511
      int key  = flat >> 3;
      int c    = (flat & 7) << 3;
      *(v8bf*)&Ks[key][c] = *(const v8bf*)&Kb[(size_t)(kb + key) * rs + c];
      v8bf vv = *(const v8bf*)&Vb[(size_t)(kb + key) * rs + c];
      #pragma unroll
      for (int e = 0; e < 8; e++) Vs[c + e][key] = vv[e];
    }
    __syncthreads();

    // S = scale * Q K^T  (+ mask bias)
    FragC s[4];
    #pragma unroll
    for (int t = 0; t < 4; t++) {
      FragC cc;
      #pragma unroll
      for (int r = 0; r < 8; r++) cc.f[r] = 0.0f;
      #pragma unroll
      for (int c = 0; c < 2; c++) {
        FragAB kf;
        const bf16_t* p = &Ks[t * 16 + l16][c * 32];
        kf.h[0] = *(const v8bf*)(p + 8 * half);
        kf.h[1] = *(const v8bf*)(p + 16 + 8 * half);
        cc.v = wmma_bf16(qa[c].v, kf.v, cc.v);
      }
      float mbias = (mb[kb + t * 16 + l16] != 0) ? 0.0f : -1e30f;
      #pragma unroll
      for (int r = 0; r < 8; r++) cc.f[r] = cc.f[r] * 0.125f + mbias;
      s[t] = cc;
    }

    // online softmax: row = half*8 + r, columns spread over 16 lanes of the half
    #pragma unroll
    for (int r = 0; r < 8; r++) {
      float x = fmaxf(fmaxf(s[0].f[r], s[1].f[r]), fmaxf(s[2].f[r], s[3].f[r]));
      #pragma unroll
      for (int off = 1; off < 16; off <<= 1) x = fmaxf(x, __shfl_xor(x, off, 32));
      float mnew = fmaxf(rm[r], x);
      float scl  = (mnew == -INFINITY) ? 1.0f : __expf(rm[r] - mnew);
      rm[r] = mnew;
      rl[r] *= scl;
      #pragma unroll
      for (int t = 0; t < 4; t++) o[t].f[r] *= scl;
    }
    #pragma unroll
    for (int t = 0; t < 4; t++)
      #pragma unroll
      for (int r = 0; r < 8; r++) {
        float p = (rm[r] == -INFINITY) ? 0.0f : __expf(s[t].f[r] - rm[r]);
        s[t].f[r] = p;
        Ps[wave][half * 8 + r][t * 16 + l16] = f2bf(p);
      }
    #pragma unroll
    for (int r = 0; r < 8; r++) {
      float x = s[0].f[r] + s[1].f[r] + s[2].f[r] + s[3].f[r];
      #pragma unroll
      for (int off = 1; off < 16; off <<= 1) x += __shfl_xor(x, off, 32);
      rl[r] += x;
    }

    // same-wave LDS RAW on Ps: make stores land before the A-frag reloads
    asm volatile("s_wait_dscnt 0x0" ::: "memory");

    // O += P V
    FragAB pa[2];
    #pragma unroll
    for (int c = 0; c < 2; c++) {
      const bf16_t* p = &Ps[wave][l16][c * 32];
      pa[c].h[0] = *(const v8bf*)(p + 8 * half);
      pa[c].h[1] = *(const v8bf*)(p + 16 + 8 * half);
    }
    #pragma unroll
    for (int t = 0; t < 4; t++)
      #pragma unroll
      for (int c = 0; c < 2; c++) {
        FragAB vf;
        const bf16_t* p = &Vs[t * 16 + l16][c * 32];
        vf.h[0] = *(const v8bf*)(p + 8 * half);
        vf.h[1] = *(const v8bf*)(p + 16 + 8 * half);
        o[t].v = wmma_bf16(pa[c].v, vf.v, o[t].v);
      }
  }

  #pragma unroll
  for (int t = 0; t < 4; t++)
    #pragma unroll
    for (int r = 0; r < 8; r++) {
      int row = qblk * 64 + wave * 16 + half * 8 + r;
      float d = rl[r];
      float v = (d > 0.0f) ? o[t].f[r] / d : 0.0f;
      Ob[(size_t)row * rs + t * 16 + l16] = f2bf(v);
    }
}

// ---------------------------------------------------------------- residual + LayerNorm
// out = LN(resid_f32 + delta_bf16) * g + b ; one block per row of 1024
__global__ void __launch_bounds__(256)
add_ln_kernel(const float* __restrict__ resid, const bf16_t* __restrict__ delta,
              const float* __restrict__ g, const float* __restrict__ b,
              float* __restrict__ outF, bf16_t* __restrict__ outB) {
  const int D = 1024;
  const int row = blockIdx.x, tid = threadIdx.x;
  __shared__ float sred[8];
  float v[4]; float s = 0.0f;
  #pragma unroll
  for (int i = 0; i < 4; i++) {
    int idx = tid + i * 256;
    v[i] = resid[(size_t)row * D + idx] + (float)delta[(size_t)row * D + idx];
    s += v[i];
  }
  #pragma unroll
  for (int off = 16; off; off >>= 1) s += __shfl_xor(s, off, 32);
  if ((tid & 31) == 0) sred[tid >> 5] = s;
  __syncthreads();
  float tot = 0.0f;
  #pragma unroll
  for (int w = 0; w < 8; w++) tot += sred[w];
  float mu = tot * (1.0f / D);

  float s2 = 0.0f;
  #pragma unroll
  for (int i = 0; i < 4; i++) { float d = v[i] - mu; s2 += d * d; }
  #pragma unroll
  for (int off = 16; off; off >>= 1) s2 += __shfl_xor(s2, off, 32);
  __syncthreads();
  if ((tid & 31) == 0) sred[tid >> 5] = s2;
  __syncthreads();
  float tv = 0.0f;
  #pragma unroll
  for (int w = 0; w < 8; w++) tv += sred[w];
  float inv = rsqrtf(tv * (1.0f / D) + 1e-5f);

  #pragma unroll
  for (int i = 0; i < 4; i++) {
    int idx = tid + i * 256;
    float y = (v[i] - mu) * inv * g[idx] + b[idx];
    if (outF) outF[(size_t)row * D + idx] = y;
    if (outB) outB[(size_t)row * D + idx] = f2bf(y);
  }
}

// ---------------------------------------------------------------- host driver
extern "C" void kernel_launch(void* const* d_in, const int* in_sizes, int n_in,
                              void* d_out, int out_size, void* d_ws, size_t ws_size,
                              hipStream_t stream) {
  (void)in_sizes; (void)n_in; (void)out_size; (void)ws_size;
  const int D = 1024, DFF = 4096, T = 2048, BATCH = 2, H = 16;
  const size_t MT = (size_t)BATCH * T;   // 4096 tokens

  const float* x   = (const float*)d_in[0];
  const int*   msk = (const int*)  d_in[1];
  const float* Wqe = (const float*)d_in[2];  const float* bqe = (const float*)d_in[3];
  const float* Wke = (const float*)d_in[4];  const float* bke = (const float*)d_in[5];
  const float* Wve = (const float*)d_in[6];  const float* bve = (const float*)d_in[7];
  const float* Wq  = (const float*)d_in[8];  const float* bq  = (const float*)d_in[9];
  const float* Wk  = (const float*)d_in[10]; const float* bk  = (const float*)d_in[11];
  const float* Wv  = (const float*)d_in[12]; const float* bv  = (const float*)d_in[13];
  const float* Wo  = (const float*)d_in[14]; const float* bo  = (const float*)d_in[15];
  const float* g1  = (const float*)d_in[16]; const float* b1  = (const float*)d_in[17];
  const float* g2  = (const float*)d_in[18]; const float* b2  = (const float*)d_in[19];
  const float* W1  = (const float*)d_in[20]; const float* bb1 = (const float*)d_in[21];
  const float* W2  = (const float*)d_in[22]; const float* bb2 = (const float*)d_in[23];
  float* out = (float*)d_out;

  char* ws = (char*)d_ws;
  size_t off = 0;
  auto alloc = [&](size_t bytes) -> char* {
    char* p = ws + off; off += (bytes + 255) & ~(size_t)255; return p;
  };
  bf16_t* xbf  = (bf16_t*)alloc(MT * D * 2);
  bf16_t* wqeT = (bf16_t*)alloc((size_t)D * D * 2);
  bf16_t* wkeT = (bf16_t*)alloc((size_t)D * D * 2);
  bf16_t* wveT = (bf16_t*)alloc((size_t)D * D * 2);
  bf16_t* wqT  = (bf16_t*)alloc((size_t)D * D * 2);
  bf16_t* wkT  = (bf16_t*)alloc((size_t)D * D * 2);
  bf16_t* wvT  = (bf16_t*)alloc((size_t)D * D * 2);
  bf16_t* woT  = (bf16_t*)alloc((size_t)D * D * 2);
  bf16_t* w1T  = (bf16_t*)alloc((size_t)D * DFF * 2);
  bf16_t* w2T  = (bf16_t*)alloc((size_t)DFF * D * 2);
  bf16_t* ebf  = (bf16_t*)alloc(MT * D * 2);
  bf16_t* qbf  = (bf16_t*)alloc(MT * D * 2);
  bf16_t* kbf  = (bf16_t*)alloc(MT * D * 2);
  bf16_t* vbf  = (bf16_t*)alloc(MT * D * 2);
  bf16_t* abf  = (bf16_t*)alloc(MT * D * 2);
  bf16_t* mhab = (bf16_t*)alloc(MT * D * 2);
  bf16_t* y1bf = (bf16_t*)alloc(MT * D * 2);
  bf16_t* hbf  = (bf16_t*)alloc(MT * DFF * 2);
  bf16_t* fbf  = (bf16_t*)alloc(MT * D * 2);
  float*  y1f  = (float*) alloc(MT * D * 4);

  // conversions
  {
    int n = (int)(MT * D);
    cvt_f32_to_bf16<<<(n + 255) / 256, 256, 0, stream>>>(x, xbf, n);
    int nd = D * D;
    cvt_transpose_f32_to_bf16<<<(nd + 255) / 256, 256, 0, stream>>>(Wqe, wqeT, D, D);
    cvt_transpose_f32_to_bf16<<<(nd + 255) / 256, 256, 0, stream>>>(Wke, wkeT, D, D);
    cvt_transpose_f32_to_bf16<<<(nd + 255) / 256, 256, 0, stream>>>(Wve, wveT, D, D);
    cvt_transpose_f32_to_bf16<<<(nd + 255) / 256, 256, 0, stream>>>(Wq, wqT, D, D);
    cvt_transpose_f32_to_bf16<<<(nd + 255) / 256, 256, 0, stream>>>(Wk, wkT, D, D);
    cvt_transpose_f32_to_bf16<<<(nd + 255) / 256, 256, 0, stream>>>(Wv, wvT, D, D);
    cvt_transpose_f32_to_bf16<<<(nd + 255) / 256, 256, 0, stream>>>(Wo, woT, D, D);
    int nf = D * DFF;
    cvt_transpose_f32_to_bf16<<<(nf + 255) / 256, 256, 0, stream>>>(W1, w1T, D, DFF);
    cvt_transpose_f32_to_bf16<<<(nf + 255) / 256, 256, 0, stream>>>(W2, w2T, DFF, D);
  }

  dim3 blk(256);
  dim3 gD(D / BN, (int)(MT / BM));    // N=1024 GEMMs
  dim3 gF(DFF / BN, (int)(MT / BM));  // N=4096 GEMM

  // q/k/v embed + projection chains (reuse ebf)
  gemm_bf16_wmma<<<gD, blk, 0, stream>>>(xbf, wqeT, bqe, ebf, (int)MT, D, D, 0);
  gemm_bf16_wmma<<<gD, blk, 0, stream>>>(ebf, wqT,  bq,  qbf, (int)MT, D, D, 0);
  gemm_bf16_wmma<<<gD, blk, 0, stream>>>(xbf, wkeT, bke, ebf, (int)MT, D, D, 0);
  gemm_bf16_wmma<<<gD, blk, 0, stream>>>(ebf, wkT,  bk,  kbf, (int)MT, D, D, 0);
  gemm_bf16_wmma<<<gD, blk, 0, stream>>>(xbf, wveT, bve, ebf, (int)MT, D, D, 0);
  gemm_bf16_wmma<<<gD, blk, 0, stream>>>(ebf, wvT,  bv,  vbf, (int)MT, D, D, 0);

  // attention
  flash_attn_wmma<<<dim3(T / 64, H, BATCH), dim3(128), 0, stream>>>(qbf, kbf, vbf, msk, abf, T, H);

  // output projection + LN1
  gemm_bf16_wmma<<<gD, blk, 0, stream>>>(abf, woT, bo, mhab, (int)MT, D, D, 0);
  add_ln_kernel<<<(int)MT, 256, 0, stream>>>(x, mhab, g1, b1, y1f, y1bf);

  // FFN + LN2
  gemm_bf16_wmma<<<gF, blk, 0, stream>>>(y1bf, w1T, bb1, hbf, (int)MT, DFF, D, 1);
  gemm_bf16_wmma<<<gD, blk, 0, stream>>>(hbf,  w2T, bb2, fbf, (int)MT, D, DFF, 0);
  add_ln_kernel<<<(int)MT, 256, 0, stream>>>(y1f, fbf, g2, b2, out, (bf16_t*)nullptr);
}
